// MultiHeadSelfAttention_87342454932016
// MI455X (gfx1250) — matrix operand order repaired and checked by reading, not compiled, not run on hardware
//
#include <hip/hip_runtime.h>
#include <hip/hip_bf16.h>

// ---------------------------------------------------------------------------
// MI455X (gfx1250) multi-head self-attention forward, bf16 WMMA pipeline.
// ---------------------------------------------------------------------------

typedef __attribute__((ext_vector_type(16))) __bf16 bf16x16;
typedef __attribute__((ext_vector_type(8)))  float  f32x8;

constexpr int NB   = 4;      // batch
constexpr int TS   = 2048;   // seq len
constexpr int DM   = 1024;   // d_model
constexpr int NH   = 16;     // heads
constexpr int DH   = 64;     // head dim
constexpr int NQKV = 3 * DM; // 3072
constexpr int MROWS = NB * TS; // 8192

// LDS row strides (halfwords), padded for conflict-free bank access,
// multiples of 8 to preserve 16B alignment of fragment reads.
constexpr int LDA_S = 40;   // A tile row stride (32 used)
constexpr int LDB_S = 40;   // B tile (transposed) row stride (32 used)
constexpr int LDP_S = 72;   // P tile row stride in flash kernel (64 used)

__device__ __forceinline__ unsigned short f2bf(float f) {
  unsigned int u = __float_as_uint(f);
  u += 0x7FFFu + ((u >> 16) & 1u);        // round to nearest even
  return (unsigned short)(u >> 16);
}
__device__ __forceinline__ float bf2f(unsigned short h) {
  return __uint_as_float(((unsigned int)h) << 16);
}

union FragBF {
  int4 i[2];
  bf16x16 v;
  unsigned short u[16];
};

// ---------------------------------------------------------------------------
// f32 -> bf16 conversion
// ---------------------------------------------------------------------------
__global__ __launch_bounds__(256)
void cvt_f32_bf16(const float* __restrict__ in, unsigned short* __restrict__ out, int n) {
  int i = blockIdx.x * blockDim.x + threadIdx.x;
  int stride = gridDim.x * blockDim.x;
  for (; i < n; i += stride) out[i] = f2bf(in[i]);
}

// ---------------------------------------------------------------------------
// Shared 128x64-tile bf16 GEMM core: C(128x64) = A(128xK) * B(KxN) slice.
// 256 threads = 8 waves; wave w computes rows [16w,16w+16) x 64 cols
// as 4 WMMA accumulators. K loop steps 32 (one 16x16x32 bf16 WMMA per tile).
// ---------------------------------------------------------------------------
__device__ __forceinline__ void gemm_core(const unsigned short* __restrict__ Abf,
                                          const unsigned short* __restrict__ Bbf,
                                          int ldb, int row0, int n0,
                                          unsigned short* As, unsigned short* Bs,
                                          f32x8 acc[4]) {
  const int tid  = threadIdx.x;
  const int wave = tid >> 5;
  const int lane = tid & 31;
  const int l16  = lane & 15;
  const int lhalf = lane >> 4;

  for (int k0 = 0; k0 < DM; k0 += 32) {
    // --- stage A tile 128x32 (row-major) ---
    {
      int r = tid >> 1, half = tid & 1;
      const int4* src = (const int4*)&Abf[(size_t)(row0 + r) * DM + k0 + half * 16];
      int4* dst = (int4*)&As[r * LDA_S + half * 16];
      dst[0] = src[0];
      dst[1] = src[1];
      if (k0 + 32 < DM)
        __builtin_prefetch(&Abf[(size_t)(row0 + r) * DM + k0 + 32 + half * 16], 0, 1);
    }
    // --- stage B tile 32x64 transposed into Bs[n][k] ---
    {
      int k = tid & 31, n8 = tid >> 5;
      int4 vv = *(const int4*)&Bbf[(size_t)(k0 + k) * ldb + n0 + n8 * 8];
      const unsigned short* pv = (const unsigned short*)&vv;
#pragma unroll
      for (int j = 0; j < 8; j++) Bs[(n8 * 8 + j) * LDB_S + k] = pv[j];
    }
    __syncthreads();

    // A fragment: 16x32 bf16, lane holds row M = lane%16,
    // element e -> Klocal = e + (lane>=16 ? 16 : 0)
    FragBF a;
    {
      int arow = wave * 16 + l16;
      int koff = lhalf * 16;
      a.i[0] = *(const int4*)&As[arow * LDA_S + koff];
      a.i[1] = *(const int4*)&As[arow * LDA_S + 8 + koff];
    }
#pragma unroll
    for (int c = 0; c < 4; c++) {
      // B fragment: 32x16 bf16, lane holds col N = lane%16,
      // element e -> Klocal = e + (lane>=16 ? 16 : 0)  (contiguous in Bs[n][k])
      FragBF b;
      int n = c * 16 + l16;
      int koff2 = lhalf * 16;
      b.i[0] = *(const int4*)&Bs[n * LDB_S + koff2];
      b.i[1] = *(const int4*)&Bs[n * LDB_S + koff2 + 8];
      acc[c] = __builtin_amdgcn_wmma_f32_16x16x32_bf16(
          false, a.v, false, b.v, (short)0, acc[c], false, false);
    }
    __syncthreads();
  }
}

// ---------------------------------------------------------------------------
// QKV projection GEMM, epilogue scatters into head-major Q,K and transposed V.
// grid = (3072/64, 8192/128)
// ---------------------------------------------------------------------------
__global__ __launch_bounds__(256)
void gemm_qkv(const unsigned short* __restrict__ xb,
              const unsigned short* __restrict__ wqkvb,
              unsigned short* __restrict__ Qb,
              unsigned short* __restrict__ Kb,
              unsigned short* __restrict__ Vt) {
  __shared__ unsigned short As[128 * LDA_S];
  __shared__ unsigned short Bs[64 * LDB_S];
  const int n0 = blockIdx.x * 64;
  const int row0 = blockIdx.y * 128;

  f32x8 acc[4];
  f32x8 z = {};
#pragma unroll
  for (int c = 0; c < 4; c++) acc[c] = z;

  gemm_core(xb, wqkvb, NQKV, row0, n0, As, Bs, acc);

  const int wave = threadIdx.x >> 5, lane = threadIdx.x & 31;
  const int l16 = lane & 15, lhalf = lane >> 4;
  const int which = n0 / DM;            // 0=Q 1=K 2=V (constant per block)
  const int h = (n0 % DM) / DH;         // head (constant per block)
#pragma unroll
  for (int c = 0; c < 4; c++) {
    int d = c * 16 + l16;               // dim within head, 0..63
#pragma unroll
    for (int r = 0; r < 8; r++) {
      int grow = row0 + wave * 16 + r + (lhalf << 3);
      int bb = grow >> 11;
      int tt = grow & (TS - 1);
      size_t bh = (size_t)(bb * NH + h);
      unsigned short v = f2bf(acc[c][r]);
      if (which == 0)      Qb[(bh * TS + tt) * DH + d] = v;
      else if (which == 1) Kb[(bh * TS + tt) * DH + d] = v;
      else                 Vt[(bh * DH + d) * TS + tt] = v;   // transposed
    }
  }
}

// ---------------------------------------------------------------------------
// RoPE on Q and K (bf16 in-place), pairing (d, d+32) per rotate_half.
// idx bits: [4:0]=j, [15:5]=t, [21:16]=b*NH+h, [22]=tensor select
// ---------------------------------------------------------------------------
__global__ __launch_bounds__(256)
void rope_kernel(unsigned short* __restrict__ Qb, unsigned short* __restrict__ Kb) {
  unsigned idx = blockIdx.x * 256u + threadIdx.x;
  int j   = idx & 31;
  int t   = (idx >> 5) & (TS - 1);
  int bh  = (idx >> 16) & 63;
  int sel = (idx >> 22) & 1;
  unsigned short* P = sel ? Kb : Qb;
  size_t base = ((size_t)bh * TS + t) * DH;
  float x1 = bf2f(P[base + j]);
  float x2 = bf2f(P[base + j + 32]);
  float ang = (float)t * __powf(10000.0f, -(float)(2 * j) * (1.0f / 64.0f));
  float s, c;
  __sincosf(ang, &s, &c);
  P[base + j]      = f2bf(x1 * c - x2 * s);
  P[base + j + 32] = f2bf(x2 * c + x1 * s);
}

// ---------------------------------------------------------------------------
// Flash attention (non-causal): one workgroup per (qblock, head, batch).
// 8 waves; wave owns 16 query rows x full Dh=64. Key sweep in chunks of 64.
// S = Q*K^T via WMMA (Dh reduction, 2 WMMAs per 16x16 S tile),
// online softmax with 16-lane shfl_xor row reductions,
// P re-laid out through wave-private LDS into A-fragment layout,
// O += P*V via WMMA using transposed V (contiguous B-fragment loads).
// ---------------------------------------------------------------------------
__global__ __launch_bounds__(256)
void flash_attn(const unsigned short* __restrict__ Qb,
                const unsigned short* __restrict__ Kb,
                const unsigned short* __restrict__ Vt,
                unsigned short* __restrict__ AO) {
  __shared__ unsigned short Plds[8 * 16 * LDP_S];
  const int tid = threadIdx.x, wave = tid >> 5, lane = tid & 31;
  const int l16 = lane & 15, lhalf = lane >> 4;
  const int qb = blockIdx.x;   // 0..15
  const int h  = blockIdx.y;   // 0..15
  const int b  = blockIdx.z;   // 0..3
  const size_t bh = (size_t)(b * NH + h);

  // Q fragments (loop-invariant): kk = 0/1 covers Dh 0..31 / 32..63
  FragBF qf[2];
  {
    int qrow = qb * 128 + wave * 16 + l16;
    const unsigned short* qptr = &Qb[(bh * TS + qrow) * DH];
#pragma unroll
    for (int kk = 0; kk < 2; kk++) {
      int d0 = lhalf * 16 + kk * 32;
      qf[kk].i[0] = *(const int4*)&qptr[d0];
      qf[kk].i[1] = *(const int4*)&qptr[d0 + 8];
    }
  }

  f32x8 O[4];
  f32x8 z = {};
#pragma unroll
  for (int dt = 0; dt < 4; dt++) O[dt] = z;
  float run_m[8], run_l[8];
#pragma unroll
  for (int r = 0; r < 8; r++) { run_m[r] = -1e30f; run_l[r] = 0.0f; }

  unsigned short* Pw = &Plds[wave * 16 * LDP_S];

  for (int kb = 0; kb < TS; kb += 64) {
    // ---- S = scale * Q K^T  (4 tiles of 16q x 16k) ----
    f32x8 S[4];
#pragma unroll
    for (int ct = 0; ct < 4; ct++) {
      f32x8 s = {};
      int key = kb + ct * 16 + l16;
      const unsigned short* kptr = &Kb[(bh * TS + key) * DH];
#pragma unroll
      for (int kk = 0; kk < 2; kk++) {
        FragBF bfr;                      // B frag: elem e -> d = e + lhalf*16 + kk*32
        int d0 = lhalf * 16 + kk * 32;
        bfr.i[0] = *(const int4*)&kptr[d0];
        bfr.i[1] = *(const int4*)&kptr[d0 + 8];
        s = __builtin_amdgcn_wmma_f32_16x16x32_bf16(
            false, qf[kk].v, false, bfr.v, (short)0, s, false, false);
      }
#pragma unroll
      for (int r = 0; r < 8; r++) s[r] *= 0.125f;   // Dh^-0.5
      S[ct] = s;
    }

    // ---- online softmax (row-wise over keys) ----
    float alpha[8];
#pragma unroll
    for (int r = 0; r < 8; r++) {
      float cm = fmaxf(fmaxf(S[0][r], S[1][r]), fmaxf(S[2][r], S[3][r]));
      for (int m = 1; m < 16; m <<= 1) cm = fmaxf(cm, __shfl_xor(cm, m, 32));
      float nm = fmaxf(run_m[r], cm);
      alpha[r] = __expf(run_m[r] - nm);
      run_m[r] = nm;
      float rs = 0.0f;
#pragma unroll
      for (int ct = 0; ct < 4; ct++) {
        float p = __expf(S[ct][r] - nm);
        S[ct][r] = p;
        rs += p;
      }
      for (int m = 1; m < 16; m <<= 1) rs += __shfl_xor(rs, m, 32);
      run_l[r] = run_l[r] * alpha[r] + rs;
    }

    // ---- spill P (bf16) to wave-private LDS; rescale O ----
#pragma unroll
    for (int ct = 0; ct < 4; ct++)
#pragma unroll
      for (int r = 0; r < 8; r++) {
        int mrow = r + lhalf * 8;
        Pw[mrow * LDP_S + ct * 16 + l16] = f2bf(S[ct][r]);
      }
#pragma unroll
    for (int dt = 0; dt < 4; dt++)
#pragma unroll
      for (int r = 0; r < 8; r++) O[dt][r] *= alpha[r];

    asm volatile("s_wait_dscnt 0" ::: "memory");  // P writes visible to our reads

    // ---- O += P * V ----
#pragma unroll
    for (int kk = 0; kk < 2; kk++) {
      FragBF pf;                          // A frag of P: row = lane%16
      {
        int k0f = lhalf * 16 + kk * 32;
        pf.i[0] = *(const int4*)&Pw[l16 * LDP_S + k0f];
        pf.i[1] = *(const int4*)&Pw[l16 * LDP_S + k0f + 8];
      }
#pragma unroll
      for (int dt = 0; dt < 4; dt++) {
        int d = dt * 16 + l16;
        const unsigned short* vptr = &Vt[(bh * DH + d) * TS];
        FragBF vf;                        // B frag: elem e -> key = kb + e + lhalf*16 + kk*32
        int key0 = kb + lhalf * 16 + kk * 32;
        vf.i[0] = *(const int4*)&vptr[key0];
        vf.i[1] = *(const int4*)&vptr[key0 + 8];
        O[dt] = __builtin_amdgcn_wmma_f32_16x16x32_bf16(
            false, pf.v, false, vf.v, (short)0, O[dt], false, false);
      }
    }
  }

  // ---- finalize: O / l, write bf16 into AO [B*T, 1024] ----
  {
    int qrow0 = qb * 128 + wave * 16;
#pragma unroll
    for (int dt = 0; dt < 4; dt++) {
      int d = dt * 16 + l16;
#pragma unroll
      for (int r = 0; r < 8; r++) {
        int q = qrow0 + r + lhalf * 8;
        float v = O[dt][r] / run_l[r];
        AO[((size_t)(b * TS + q)) * DM + h * DH + d] = f2bf(v);
      }
    }
  }
}

// ---------------------------------------------------------------------------
// Output projection GEMM: AO(8192x1024) * w_out(1024x1024) -> f32 out.
// grid = (1024/64, 8192/128)
// ---------------------------------------------------------------------------
__global__ __launch_bounds__(256)
void gemm_out(const unsigned short* __restrict__ AO,
              const unsigned short* __restrict__ woutb,
              float* __restrict__ Cout) {
  __shared__ unsigned short As[128 * LDA_S];
  __shared__ unsigned short Bs[64 * LDB_S];
  const int n0 = blockIdx.x * 64;
  const int row0 = blockIdx.y * 128;

  f32x8 acc[4];
  f32x8 z = {};
#pragma unroll
  for (int c = 0; c < 4; c++) acc[c] = z;

  gemm_core(AO, woutb, DM, row0, n0, As, Bs, acc);

  const int wave = threadIdx.x >> 5, lane = threadIdx.x & 31;
  const int l16 = lane & 15, lhalf = lane >> 4;
#pragma unroll
  for (int c = 0; c < 4; c++) {
    int col = n0 + c * 16 + l16;
#pragma unroll
    for (int r = 0; r < 8; r++) {
      int grow = row0 + wave * 16 + r + (lhalf << 3);
      Cout[(size_t)grow * DM + col] = acc[c][r];
    }
  }
}

// ---------------------------------------------------------------------------
// Host-side launch
// ---------------------------------------------------------------------------
extern "C" void kernel_launch(void* const* d_in, const int* in_sizes, int n_in,
                              void* d_out, int out_size, void* d_ws, size_t ws_size,
                              hipStream_t stream) {
  const float* x     = (const float*)d_in[0];
  const float* w_qkv = (const float*)d_in[1];
  const float* w_out = (const float*)d_in[2];
  float* out = (float*)d_out;

  char* ws = (char*)d_ws;
  // workspace layout (bytes)
  constexpr size_t OFF_XB    = 0;                        // 8192*1024 bf16 = 16 MiB
  constexpr size_t OFF_WQKV  = OFF_XB    + (size_t)MROWS * DM * 2;
  constexpr size_t OFF_WOUT  = OFF_WQKV  + (size_t)DM * NQKV * 2;
  constexpr size_t OFF_Q     = OFF_WOUT  + (size_t)DM * DM * 2;
  constexpr size_t OFF_K     = OFF_Q     + (size_t)MROWS * DM * 2;
  constexpr size_t OFF_V     = OFF_K     + (size_t)MROWS * DM * 2;
  constexpr size_t OFF_AO    = OFF_V     + (size_t)MROWS * DM * 2;

  unsigned short* xb    = (unsigned short*)(ws + OFF_XB);
  unsigned short* wqkvb = (unsigned short*)(ws + OFF_WQKV);
  unsigned short* woutb = (unsigned short*)(ws + OFF_WOUT);
  unsigned short* Qb    = (unsigned short*)(ws + OFF_Q);
  unsigned short* Kb    = (unsigned short*)(ws + OFF_K);
  unsigned short* Vt    = (unsigned short*)(ws + OFF_V);
  unsigned short* AO    = (unsigned short*)(ws + OFF_AO);

  cvt_f32_bf16<<<4096, 256, 0, stream>>>(x, xb, MROWS * DM);
  cvt_f32_bf16<<<2048, 256, 0, stream>>>(w_qkv, wqkvb, DM * NQKV);
  cvt_f32_bf16<<<1024, 256, 0, stream>>>(w_out, woutb, DM * DM);

  gemm_qkv<<<dim3(NQKV / 64, MROWS / 128), 256, 0, stream>>>(xb, wqkvb, Qb, Kb, Vt);

  rope_kernel<<<(2u * NB * NH * TS * 32) / 256, 256, 0, stream>>>(Qb, Kb);

  flash_attn<<<dim3(TS / 128, NH, NB), 256, 0, stream>>>(Qb, Kb, Vt, AO);

  gemm_out<<<dim3(DM / 64, MROWS / 128), 256, 0, stream>>>(AO, woutb, out);
}
